// SSD_44478681317686
// MI455X (gfx1250) — matrix operand order repaired and checked by reading, not compiled, not run on hardware
//
#include <hip/hip_runtime.h>
#include <hip/hip_bf16.h>
#include <stdint.h>

// ---------------- problem constants (match reference) ----------------
#define D_MODEL   1024
#define D_STATE   128
#define D_CONV    4
#define D_INNER   2048
#define NHEADS    32
#define HEADDIM   64
#define CONV_DIM  2304           // D_INNER + 2*D_STATE
#define D_IN_PROJ 4384           // 2*D_INNER + 2*D_STATE + NHEADS
#define ZXS       4480           // D_IN_PROJ padded to a multiple of 128 (35*128)
#define CHUNK     64
#define BATCH     2
#define SEQ       4096
#define ROWS      (BATCH*SEQ)    // 8192
#define NCHUNK    (SEQ/CHUNK)    // 64
#define NBC       (BATCH*NCHUNK) // 128
#define D_FF      4096
#define EPS_F     1e-6f

// ---------------- types ----------------
typedef __bf16 bf16_t;
typedef __attribute__((ext_vector_type(16))) __bf16 v16bf;
typedef __attribute__((ext_vector_type(8)))  float  v8f;

struct __align__(16) U128 { unsigned int a, b, c, d; };
union Frag { v16bf v; U128 q[2]; };

__device__ __forceinline__ bf16_t f2bf(float f) {
  union { float f; unsigned int u; } x; x.f = f;
  unsigned int r = x.u + 0x7FFFu + ((x.u >> 16) & 1u);   // RNE
  union { unsigned short s; bf16_t b; } y; y.s = (unsigned short)(r >> 16);
  return y.b;
}
__device__ __forceinline__ float bf2f(bf16_t b) {
  union { unsigned short s; bf16_t b; } x; x.b = b;
  union { float f; unsigned int u; } y; y.u = ((unsigned int)x.s) << 16;
  return y.f;
}
__device__ __forceinline__ float siluf(float x) { return x / (1.0f + expf(-x)); }

#define WMMA_BF16(A, B, C) \
  __builtin_amdgcn_wmma_f32_16x16x32_bf16(false, (A), false, (B), (short)0, (C), false, false)

// ---------------- CDNA5 async global->LDS (ASYNCcnt) ----------------
// Generic pointer to a __shared__ object: high 32 bits = LDS aperture,
// low 32 bits = wave-relative LDS byte address (ISA 10.2 aperture rules).
__device__ __forceinline__ unsigned lds_addr(const void* p) {
  return (unsigned)(uintptr_t)p;
}
// GLOBAL_LOAD_ASYNC_TO_LDS_B128: VDST = LDS byte-address VGPR, VADDR = 64b global addr (GV mode)
__device__ __forceinline__ void async_ld_b128(unsigned lds_off, const void* gptr) {
  asm volatile("global_load_async_to_lds_b128 %0, %1, off"
               :: "v"(lds_off), "v"((unsigned long long)(uintptr_t)gptr)
               : "memory");
}
__device__ __forceinline__ void wait_async0() {
  asm volatile("s_wait_asynccnt 0x0" ::: "memory");
}

// ===================================================================
// Tiled bf16 GEMM:  Out[m,n] = sum_k A[m*K+k] * W[n*K+k]  (+ optional f32 residual)
// Requirements (guaranteed by caller): M % 128 == 0, N % 128 == 0, K % 32 == 0.
// Block tile 128x128x32, 8 waves (4x2), each wave: 2x4 WMMA accumulators.
// Branch-free double-buffered async global->LDS staging (ASYNCcnt DMA).
// ===================================================================
#define BM 128
#define BN 128
#define BK 32
#define LDT 48   // LDS row stride in halves (96B, 16B-aligned, staggered)

__global__ __launch_bounds__(256)
void gemm_bf16_wmma(const bf16_t* __restrict__ A, const bf16_t* __restrict__ W,
                    float* __restrict__ Out, const float* __restrict__ Res,
                    int M, int N, int K)
{
  __shared__ __align__(16) bf16_t As[2][BM * LDT];
  __shared__ __align__(16) bf16_t Bs[2][BM * LDT];
  const int tid  = threadIdx.x;
  const int bm   = blockIdx.y * BM;
  const int bn   = blockIdx.x * BN;
  const int wave = tid >> 5;
  const int lane = tid & 31;
  const int wm   = wave >> 1;    // 0..3  (M direction)
  const int wn   = wave & 1;     // 0..1  (N direction)
  const int lrow = lane & 15;
  const int lhal = lane >> 4;

  const v8f vz = {0.f,0.f,0.f,0.f,0.f,0.f,0.f,0.f};
  v8f acc[2][4];
#pragma unroll
  for (int i = 0; i < 2; ++i)
#pragma unroll
    for (int j = 0; j < 4; ++j) acc[i][j] = vz;

  // cooperative staging: thread -> (row, 16-half segment); 32B per thread per tile
  const int lr = tid >> 1, ls = tid & 1;
  const bf16_t* aptr = A + (size_t)(bm + lr) * K + ls * 16;
  const bf16_t* bptr = W + (size_t)(bn + lr) * K + ls * 16;
  const unsigned aoff[2] = { lds_addr(&As[0][lr*LDT + ls*16]), lds_addr(&As[1][lr*LDT + ls*16]) };
  const unsigned boff[2] = { lds_addr(&Bs[0][lr*LDT + ls*16]), lds_addr(&Bs[1][lr*LDT + ls*16]) };

  auto stage = [&](int k0, int buf) {
    async_ld_b128(aoff[buf],      aptr + k0);
    async_ld_b128(aoff[buf] + 16, aptr + k0 + 8);
    async_ld_b128(boff[buf],      bptr + k0);
    async_ld_b128(boff[buf] + 16, bptr + k0 + 8);
  };

  stage(0, 0);
  wait_async0();
  __syncthreads();

  int buf = 0;
  for (int k0 = 0; k0 < K; k0 += BK) {
    const int nbuf = buf ^ 1;
    if (k0 + BK < K) stage(k0 + BK, nbuf);   // DMA next tile while computing

    Frag a[2], bf_[4];
#pragma unroll
    for (int i = 0; i < 2; ++i) {
      const int rr = wm*32 + i*16 + lrow;
      a[i].q[0] = *(const U128*)&As[buf][rr*LDT + lhal*8];
      a[i].q[1] = *(const U128*)&As[buf][rr*LDT + 16 + lhal*8];
    }
#pragma unroll
    for (int j = 0; j < 4; ++j) {
      const int cc = wn*64 + j*16 + lrow;
      bf_[j].q[0] = *(const U128*)&Bs[buf][cc*LDT + lhal*8];
      bf_[j].q[1] = *(const U128*)&Bs[buf][cc*LDT + 16 + lhal*8];
    }
#pragma unroll
    for (int i = 0; i < 2; ++i)
#pragma unroll
      for (int j = 0; j < 4; ++j)
        acc[i][j] = WMMA_BF16(a[i].v, bf_[j].v, acc[i][j]);

    wait_async0();      // my async stores to LDS landed
    __syncthreads();    // everyone's landed; safe to read nbuf / overwrite buf
    buf = nbuf;
  }

#pragma unroll
  for (int i = 0; i < 2; ++i)
#pragma unroll
    for (int j = 0; j < 4; ++j) {
      const int col = bn + wn*64 + j*16 + lrow;
      const int rowb = bm + wm*32 + i*16 + 8*lhal;
#pragma unroll
      for (int e = 0; e < 8; ++e) {
        const size_t o = (size_t)(rowb + e) * (size_t)N + col;
        float v = acc[i][j][e];
        if (Res) v += Res[o];
        Out[o] = v;
      }
    }
}

// ===================================================================
// RMSNorm (f32 in, bf16 out): one block per row
// ===================================================================
__global__ __launch_bounds__(256)
void rmsnorm_bf16_kernel(const float* __restrict__ in, const float* __restrict__ w,
                         bf16_t* __restrict__ out, int cols)
{
  const int row = blockIdx.x, tid = threadIdx.x;
  const float* x = in + (size_t)row * cols;
  float ss = 0.f;
  for (int c = tid; c < cols; c += 256) { float v = x[c]; ss += v * v; }
  __shared__ float red[256];
  red[tid] = ss; __syncthreads();
  for (int s2 = 128; s2 > 0; s2 >>= 1) { if (tid < s2) red[tid] += red[tid + s2]; __syncthreads(); }
  const float scale = rsqrtf(red[0] / (float)cols + EPS_F);
  bf16_t* o = out + (size_t)row * cols;
  for (int c = tid; c < cols; c += 256) o[c] = f2bf(x[c] * scale * w[c]);
}

// ===================================================================
// f32 -> bf16 conversion (weights), grid-stride
// ===================================================================
__global__ void cvt_bf16_kernel(const float* __restrict__ in, bf16_t* __restrict__ out, size_t n)
{
  size_t i = (size_t)blockIdx.x * blockDim.x + threadIdx.x;
  const size_t stride = (size_t)gridDim.x * blockDim.x;
  for (; i < n; i += stride) out[i] = f2bf(in[i]);
}

// ===================================================================
// dt = softplus(dt_raw + dt_bias); dtA = -exp(A_log)*dt
// ===================================================================
__global__ void dt_kernel(const float* __restrict__ zx, const float* __restrict__ dt_bias,
                          const float* __restrict__ A_log,
                          float* __restrict__ DT, float* __restrict__ DTA)
{
  const int idx = blockIdx.x * 256 + threadIdx.x;     // ROWS*NHEADS
  const int row = idx >> 5, h = idx & 31;
  const float v = zx[(size_t)row * ZXS + (D_INNER + CONV_DIM) + h] + dt_bias[h];
  const float sp = (v > 20.f) ? v : log1pf(expf(v));
  DT[idx]  = sp;
  DTA[idx] = -expf(A_log[h]) * sp;
}

// ===================================================================
// Causal depthwise conv(4) + bias + silu, then route into x / B / C;
// x is stored both raw (XS) and pre-scaled by dt (XDT), as bf16.
// ===================================================================
__global__ __launch_bounds__(256)
void conv_silu_kernel(const float* __restrict__ zx, const float* __restrict__ conv_w,
                      const float* __restrict__ conv_b, const float* __restrict__ DT,
                      bf16_t* __restrict__ XS, bf16_t* __restrict__ XDT,
                      bf16_t* __restrict__ Bm, bf16_t* __restrict__ Cm)
{
  const int row = blockIdx.x;
  const int b = row >> 12;           // row / SEQ
  const int t = row & (SEQ - 1);
  for (int ch = threadIdx.x; ch < CONV_DIM; ch += 256) {
    float acc = conv_b[ch];
#pragma unroll
    for (int k = 0; k < D_CONV; ++k) {
      const int tt = t - (D_CONV - 1) + k;
      if (tt >= 0)
        acc += conv_w[ch * D_CONV + k] * zx[(size_t)(b * SEQ + tt) * ZXS + D_INNER + ch];
    }
    const float sv = siluf(acc);
    if (ch < D_INNER) {
      const int h = ch >> 6;
      XS [(size_t)row * D_INNER + ch] = f2bf(sv);
      XDT[(size_t)row * D_INNER + ch] = f2bf(sv * DT[row * NHEADS + h]);
    } else if (ch < D_INNER + D_STATE) {
      Bm[(size_t)row * D_STATE + (ch - D_INNER)] = f2bf(sv);
    } else {
      Cm[(size_t)row * D_STATE + (ch - D_INNER - D_STATE)] = f2bf(sv);
    }
  }
}

// ===================================================================
// Per-chunk inclusive cumsum of dtA over l (64) for each (b,c,h)
// ===================================================================
__global__ __launch_bounds__(64)
void acs_kernel(const float* __restrict__ DTA, float* __restrict__ ACS, float* __restrict__ CHS)
{
  const int bch = blockIdx.x;               // bc*NHEADS + h
  const int bc = bch >> 5, h = bch & 31;
  const int l = threadIdx.x;
  __shared__ float sbuf[CHUNK];
  sbuf[l] = DTA[(size_t)(bc * CHUNK + l) * NHEADS + h];
  __syncthreads();
  for (int off = 1; off < CHUNK; off <<= 1) {
    const float v = (l >= off) ? sbuf[l - off] : 0.f;
    __syncthreads();
    sbuf[l] += v;
    __syncthreads();
  }
  ACS[(size_t)bch * CHUNK + l] = sbuf[l];
  if (l == CHUNK - 1) CHS[bch] = sbuf[l];
}

// ===================================================================
// Per-chunk states[p,n] = sum_l decay(l) * xdt[l,p] * B[l,n]   (WMMA, 64x128x64)
// ===================================================================
#define SLD 72     // 144B stride (16B aligned)
__global__ __launch_bounds__(128)
void chunk_states_kernel(const bf16_t* __restrict__ XDT, const bf16_t* __restrict__ Bm,
                         const float* __restrict__ ACS, const float* __restrict__ CHS,
                         float* __restrict__ STATES)
{
  const int bch = blockIdx.x;
  const int bc = bch >> 5, h = bch & 31;
  const int tid = threadIdx.x;
  __shared__ __align__(16) bf16_t At[CHUNK * SLD];     // [p][l]
  __shared__ __align__(16) bf16_t Bt[D_STATE * SLD];   // [n][l]
  __shared__ float dec[CHUNK];
  if (tid < CHUNK) dec[tid] = expf(CHS[bch] - ACS[(size_t)bch * CHUNK + tid]);
  __syncthreads();
  for (int idx = tid; idx < CHUNK * HEADDIM; idx += 128) {
    const int p = idx >> 6, l = idx & 63;
    At[p * SLD + l] = f2bf(bf2f(XDT[(size_t)(bc * CHUNK + l) * D_INNER + h * HEADDIM + p]) * dec[l]);
  }
  for (int idx = tid; idx < D_STATE * CHUNK; idx += 128) {
    const int n = idx >> 6, l = idx & 63;
    Bt[n * SLD + l] = Bm[(size_t)(bc * CHUNK + l) * D_STATE + n];
  }
  __syncthreads();

  const int wave = tid >> 5, lane = tid & 31, lrow = lane & 15, lhal = lane >> 4;
  const v8f vz = {0.f,0.f,0.f,0.f,0.f,0.f,0.f,0.f};
  v8f acc[8];
#pragma unroll
  for (int j = 0; j < 8; ++j) acc[j] = vz;

#pragma unroll
  for (int k0 = 0; k0 < CHUNK; k0 += 32) {
    Frag a;
    const int rr = wave * 16 + lrow;
    a.q[0] = *(const U128*)&At[rr * SLD + k0 + lhal * 8];
    a.q[1] = *(const U128*)&At[rr * SLD + k0 + 16 + lhal * 8];
#pragma unroll
    for (int j = 0; j < 8; ++j) {
      Frag bf_;
      const int cc = j * 16 + lrow;
      bf_.q[0] = *(const U128*)&Bt[cc * SLD + k0 + lhal * 8];
      bf_.q[1] = *(const U128*)&Bt[cc * SLD + k0 + 16 + lhal * 8];
      acc[j] = WMMA_BF16(a.v, bf_.v, acc[j]);
    }
  }
  float* out = STATES + (size_t)bch * (HEADDIM * D_STATE);
#pragma unroll
  for (int j = 0; j < 8; ++j) {
    const int n = j * 16 + lrow;
    const int pb = wave * 16 + 8 * lhal;
#pragma unroll
    for (int e = 0; e < 8; ++e)
      out[(size_t)(pb + e) * D_STATE + n] = acc[j][e];
  }
}

// ===================================================================
// Sequential inter-chunk scan: P_{c+1} = exp(T_c)*P_c + states_c
// State (64x128 f32) lives entirely in registers: 32 f32/lane * 256 lanes.
// Writes prev-state (P_c) as bf16 for the Y_off WMMA.
// ===================================================================
__global__ __launch_bounds__(256)
void recurrence_kernel(const float* __restrict__ STATES, const float* __restrict__ CHS,
                       bf16_t* __restrict__ PREVS)
{
  const int b = blockIdx.x >> 5, h = blockIdx.x & 31;
  const int base = threadIdx.x * 32;
  float R[32];
#pragma unroll
  for (int i = 0; i < 32; ++i) R[i] = 0.f;
  for (int c = 0; c < NCHUNK; ++c) {
    const int bch = (b * NCHUNK + c) * NHEADS + h;
    const size_t off = (size_t)bch * (HEADDIM * D_STATE) + base;
#pragma unroll
    for (int i = 0; i < 32; ++i) PREVS[off + i] = f2bf(R[i]);
    const float d = expf(CHS[bch]);
#pragma unroll
    for (int i = 0; i < 32; ++i) R[i] = R[i] * d + STATES[off + i];
  }
}

// ===================================================================
// Y kernel: per (b,c,h)
//   G  = C·Bᵀ (WMMA)  -> masked & decayed -> bf16 in LDS
//   Y1 = (L∘G)·Xdt     (WMMA)
//   Y2 = C·prevSᵀ      (WMMA, B-operand fragments straight from global)
//   Y  = Y1 + exp(acs)·Y2 + D[h]·x
// C/B chunk tiles staged via async global->LDS DMA.
// ===================================================================
#define CLD 136    // 272B stride (16B aligned)
__global__ __launch_bounds__(128)
void y_kernel(const bf16_t* __restrict__ Cm, const bf16_t* __restrict__ Bm,
              const bf16_t* __restrict__ XDT, const bf16_t* __restrict__ XS,
              const bf16_t* __restrict__ PREVS, const float* __restrict__ ACS,
              const float* __restrict__ Dp, float* __restrict__ Y)
{
  const int bch = blockIdx.x;
  const int bc = bch >> 5, h = bch & 31;
  const int tid = threadIdx.x;
  __shared__ __align__(16) bf16_t Cs [CHUNK * CLD];    // [l][n]
  __shared__ __align__(16) bf16_t Bs2[CHUNK * CLD];    // [s][n]
  __shared__ __align__(16) bf16_t Gs [CHUNK * SLD];    // [l][s]
  __shared__ __align__(16) bf16_t Xs [HEADDIM * SLD];  // [p][s]
  __shared__ float acs[CHUNK];

  // async DMA: 64 rows x 16 segments(8 halves) for each of C and B
  for (int seg = tid; seg < CHUNK * 16; seg += 128) {
    const int l = seg >> 4, s8 = (seg & 15) * 8;
    async_ld_b128(lds_addr(&Cs [l * CLD + s8]), Cm + (size_t)(bc * CHUNK + l) * D_STATE + s8);
    async_ld_b128(lds_addr(&Bs2[l * CLD + s8]), Bm + (size_t)(bc * CHUNK + l) * D_STATE + s8);
  }
  if (tid < CHUNK) acs[tid] = ACS[(size_t)bch * CHUNK + tid];
  for (int idx = tid; idx < HEADDIM * CHUNK; idx += 128) {
    const int p = idx >> 6, s = idx & 63;
    Xs[p * SLD + s] = XDT[(size_t)(bc * CHUNK + s) * D_INNER + h * HEADDIM + p];
  }
  wait_async0();
  __syncthreads();

  const int wave = tid >> 5, lane = tid & 31, lrow = lane & 15, lhal = lane >> 4;
  const v8f vz = {0.f,0.f,0.f,0.f,0.f,0.f,0.f,0.f};

  // ---- Phase 1: G = C @ B^T (K = 128)
  v8f g[4];
#pragma unroll
  for (int j = 0; j < 4; ++j) g[j] = vz;
#pragma unroll
  for (int k0 = 0; k0 < D_STATE; k0 += 32) {
    Frag a;
    const int rr = wave * 16 + lrow;
    a.q[0] = *(const U128*)&Cs[rr * CLD + k0 + lhal * 8];
    a.q[1] = *(const U128*)&Cs[rr * CLD + k0 + 16 + lhal * 8];
#pragma unroll
    for (int j = 0; j < 4; ++j) {
      Frag bf_;
      const int cc = j * 16 + lrow;
      bf_.q[0] = *(const U128*)&Bs2[cc * CLD + k0 + lhal * 8];
      bf_.q[1] = *(const U128*)&Bs2[cc * CLD + k0 + 16 + lhal * 8];
      g[j] = WMMA_BF16(a.v, bf_.v, g[j]);
    }
  }
  // mask to lower-triangular, apply segment decay, park as bf16
#pragma unroll
  for (int j = 0; j < 4; ++j) {
    const int sC = j * 16 + lrow;
    const int lb = wave * 16 + 8 * lhal;
#pragma unroll
    for (int e = 0; e < 8; ++e) {
      const int l = lb + e;
      const float v = (sC <= l) ? g[j][e] * expf(acs[l] - acs[sC]) : 0.f;
      Gs[l * SLD + sC] = f2bf(v);
    }
  }
  __syncthreads();

  // ---- Phase 2: Y1 = Gm @ Xdt (K = 64)
  v8f y1[4];
#pragma unroll
  for (int j = 0; j < 4; ++j) y1[j] = vz;
#pragma unroll
  for (int k0 = 0; k0 < CHUNK; k0 += 32) {
    Frag a;
    const int rr = wave * 16 + lrow;
    a.q[0] = *(const U128*)&Gs[rr * SLD + k0 + lhal * 8];
    a.q[1] = *(const U128*)&Gs[rr * SLD + k0 + 16 + lhal * 8];
#pragma unroll
    for (int j = 0; j < 4; ++j) {
      Frag bf_;
      const int cc = j * 16 + lrow;
      bf_.q[0] = *(const U128*)&Xs[cc * SLD + k0 + lhal * 8];
      bf_.q[1] = *(const U128*)&Xs[cc * SLD + k0 + 16 + lhal * 8];
      y1[j] = WMMA_BF16(a.v, bf_.v, y1[j]);
    }
  }

  // ---- Phase 3: Y2 = C @ prevS^T (K = 128); prevS[p][n] is already B-fragment layout
  v8f y2[4];
#pragma unroll
  for (int j = 0; j < 4; ++j) y2[j] = vz;
  const bf16_t* Pv = PREVS + (size_t)bch * (HEADDIM * D_STATE);
#pragma unroll
  for (int k0 = 0; k0 < D_STATE; k0 += 32) {
    Frag a;
    const int rr = wave * 16 + lrow;
    a.q[0] = *(const U128*)&Cs[rr * CLD + k0 + lhal * 8];
    a.q[1] = *(const U128*)&Cs[rr * CLD + k0 + 16 + lhal * 8];
#pragma unroll
    for (int j = 0; j < 4; ++j) {
      Frag bf_;
      const int pp = j * 16 + lrow;
      bf_.q[0] = *(const U128*)(Pv + (size_t)pp * D_STATE + k0 + lhal * 8);
      bf_.q[1] = *(const U128*)(Pv + (size_t)pp * D_STATE + k0 + 16 + lhal * 8);
      y2[j] = WMMA_BF16(a.v, bf_.v, y2[j]);
    }
  }

  // ---- Combine + D*x skip
  const float Dh = Dp[h];
#pragma unroll
  for (int j = 0; j < 4; ++j) {
    const int p = j * 16 + lrow;
    const int lb = wave * 16 + 8 * lhal;
#pragma unroll
    for (int e = 0; e < 8; ++e) {
      const int l = lb + e;
      const int row = bc * CHUNK + l;
      const size_t o = (size_t)row * D_INNER + h * HEADDIM + p;
      Y[o] = y1[j][e] + expf(acs[l]) * y2[j][e] + Dh * bf2f(XS[o]);
    }
  }
}

// ===================================================================
// yg = rmsnorm(y * silu(z), ssm_norm_w)  -> bf16
// ===================================================================
__global__ __launch_bounds__(256)
void gate_rms_kernel(const float* __restrict__ Y, const float* __restrict__ ZX,
                     const float* __restrict__ w, bf16_t* __restrict__ YG)
{
  const int row = blockIdx.x, tid = threadIdx.x;
  float tv[8];
  float ss = 0.f;
#pragma unroll
  for (int i = 0; i < 8; ++i) {
    const int c = tid + i * 256;
    const float z = ZX[(size_t)row * ZXS + c];
    const float t = Y[(size_t)row * D_INNER + c] * siluf(z);
    tv[i] = t; ss += t * t;
  }
  __shared__ float red[256];
  red[tid] = ss; __syncthreads();
  for (int s2 = 128; s2 > 0; s2 >>= 1) { if (tid < s2) red[tid] += red[tid + s2]; __syncthreads(); }
  const float scale = rsqrtf(red[0] / (float)D_INNER + EPS_F);
#pragma unroll
  for (int i = 0; i < 8; ++i) {
    const int c = tid + i * 256;
    YG[(size_t)row * D_INNER + c] = f2bf(tv[i] * scale * w[c]);
  }
}

// ===================================================================
// gated = silu(gate) * up  -> bf16
// ===================================================================
__global__ void silu_mul_kernel(const float* __restrict__ G, const float* __restrict__ U,
                                bf16_t* __restrict__ O, size_t n)
{
  const size_t i = (size_t)blockIdx.x * 256 + threadIdx.x;
  if (i < n) O[i] = f2bf(siluf(G[i]) * U[i]);
}

// ===================================================================
// launcher
// ===================================================================
extern "C" void kernel_launch(void* const* d_in, const int* in_sizes, int n_in,
                              void* d_out, int out_size, void* d_ws, size_t ws_size,
                              hipStream_t stream)
{
  (void)in_sizes; (void)n_in; (void)out_size; (void)ws_size;
  const float* x        = (const float*)d_in[0];
  const float* prenorm  = (const float*)d_in[1];
  const float* postnorm = (const float*)d_in[2];
  const float* inproj   = (const float*)d_in[3];
  const float* convw    = (const float*)d_in[4];
  const float* convb    = (const float*)d_in[5];
  const float* dt_bias  = (const float*)d_in[6];
  const float* A_log    = (const float*)d_in[7];
  const float* Dp       = (const float*)d_in[8];
  const float* ssmw     = (const float*)d_in[9];
  const float* outproj  = (const float*)d_in[10];
  const float* gatew    = (const float*)d_in[11];
  const float* upw      = (const float*)d_in[12];
  const float* downw    = (const float*)d_in[13];
  float* out = (float*)d_out;

  char* ws = (char*)d_ws;
  size_t off = 0;
  auto alloc = [&](size_t bytes) -> char* {
    char* p = ws + off;
    off = (off + bytes + 255) & ~(size_t)255;
    return p;
  };

  // in_proj weights padded to ZXS rows: rows [4384,4480) are junk and only feed
  // the padded output columns, which are never read downstream.
  bf16_t* Winb   = (bf16_t*)alloc((size_t)ZXS * D_MODEL * 2);
  bf16_t* Woutb  = (bf16_t*)alloc((size_t)D_MODEL * D_INNER * 2);
  bf16_t* Wgb    = (bf16_t*)alloc((size_t)D_FF * D_MODEL * 2);
  bf16_t* Wub    = (bf16_t*)alloc((size_t)D_FF * D_MODEL * 2);
  bf16_t* Wdb    = (bf16_t*)alloc((size_t)D_MODEL * D_FF * 2);
  bf16_t* XN     = (bf16_t*)alloc((size_t)ROWS * D_MODEL * 2);
  float*  ZX     = (float*) alloc((size_t)ROWS * ZXS * 4);   // later reused as GATE f32
  bf16_t* XS     = (bf16_t*)alloc((size_t)ROWS * D_INNER * 2);
  bf16_t* XDT    = (bf16_t*)alloc((size_t)ROWS * D_INNER * 2);
  bf16_t* Bm     = (bf16_t*)alloc((size_t)ROWS * D_STATE * 2);
  bf16_t* Cm     = (bf16_t*)alloc((size_t)ROWS * D_STATE * 2);
  float*  DT     = (float*) alloc((size_t)ROWS * NHEADS * 4);
  float*  DTA    = (float*) alloc((size_t)ROWS * NHEADS * 4);
  float*  ACS    = (float*) alloc((size_t)NBC * NHEADS * CHUNK * 4);
  float*  CHS    = (float*) alloc((size_t)NBC * NHEADS * 4);
  float*  STATES = (float*) alloc((size_t)NBC * NHEADS * HEADDIM * D_STATE * 4); // reused as UP f32
  bf16_t* PREVS  = (bf16_t*)alloc((size_t)NBC * NHEADS * HEADDIM * D_STATE * 2);
  float*  Yb     = (float*) alloc((size_t)ROWS * D_INNER * 4);
  bf16_t* YG     = (bf16_t*)alloc((size_t)ROWS * D_INNER * 2);
  float*  H      = (float*) alloc((size_t)ROWS * D_MODEL * 4);
  bf16_t* HN     = (bf16_t*)alloc((size_t)ROWS * D_MODEL * 2);
  bf16_t* GATED  = (bf16_t*)alloc((size_t)ROWS * D_FF * 2);
  float* GATE = ZX;       // zxbcdt fully consumed before MLP
  float* UP   = STATES;   // chunk states fully consumed before MLP

  // weights -> bf16
  cvt_bf16_kernel<<<4096, 256, 0, stream>>>(inproj,  Winb,  (size_t)D_IN_PROJ * D_MODEL);
  cvt_bf16_kernel<<<4096, 256, 0, stream>>>(outproj, Woutb, (size_t)D_MODEL * D_INNER);
  cvt_bf16_kernel<<<4096, 256, 0, stream>>>(gatew,   Wgb,   (size_t)D_FF * D_MODEL);
  cvt_bf16_kernel<<<4096, 256, 0, stream>>>(upw,     Wub,   (size_t)D_FF * D_MODEL);
  cvt_bf16_kernel<<<4096, 256, 0, stream>>>(downw,   Wdb,   (size_t)D_MODEL * D_FF);

  // prenorm -> bf16 activations
  rmsnorm_bf16_kernel<<<ROWS, 256, 0, stream>>>(x, prenorm, XN, D_MODEL);

  // in_proj GEMM: [8192,1024] x [4480(pad),1024]^T -> ZX (stride 4480)
  gemm_bf16_wmma<<<dim3(ZXS / BN, ROWS / BM), 256, 0, stream>>>(
      XN, Winb, ZX, nullptr, ROWS, ZXS, D_MODEL);

  // dt softplus + A*dt
  dt_kernel<<<(ROWS * NHEADS) / 256, 256, 0, stream>>>(ZX, dt_bias, A_log, DT, DTA);

  // conv + silu + routing
  conv_silu_kernel<<<ROWS, 256, 0, stream>>>(ZX, convw, convb, DT, XS, XDT, Bm, Cm);

  // SSD: cumsum, chunk states, inter-chunk scan, intra+inter outputs
  acs_kernel<<<NBC * NHEADS, 64, 0, stream>>>(DTA, ACS, CHS);
  chunk_states_kernel<<<NBC * NHEADS, 128, 0, stream>>>(XDT, Bm, ACS, CHS, STATES);
  recurrence_kernel<<<BATCH * NHEADS, 256, 0, stream>>>(STATES, CHS, PREVS);
  y_kernel<<<NBC * NHEADS, 128, 0, stream>>>(Cm, Bm, XDT, XS, PREVS, ACS, Dp, Yb);

  // gated rmsnorm -> bf16
  gate_rms_kernel<<<ROWS, 256, 0, stream>>>(Yb, ZX, ssmw, YG);

  // out_proj GEMM + residual x  -> H
  gemm_bf16_wmma<<<dim3(D_MODEL / BN, ROWS / BM), 256, 0, stream>>>(
      YG, Woutb, H, x, ROWS, D_MODEL, D_INNER);

  // postnorm -> bf16
  rmsnorm_bf16_kernel<<<ROWS, 256, 0, stream>>>(H, postnorm, HN, D_MODEL);

  // MLP
  gemm_bf16_wmma<<<dim3(D_FF / BN, ROWS / BM), 256, 0, stream>>>(
      HN, Wgb, GATE, nullptr, ROWS, D_FF, D_MODEL);
  gemm_bf16_wmma<<<dim3(D_FF / BN, ROWS / BM), 256, 0, stream>>>(
      HN, Wub, UP, nullptr, ROWS, D_FF, D_MODEL);
  silu_mul_kernel<<<(unsigned)(((size_t)ROWS * D_FF) / 256), 256, 0, stream>>>(
      GATE, UP, GATED, (size_t)ROWS * D_FF);
  gemm_bf16_wmma<<<dim3(D_MODEL / BN, ROWS / BM), 256, 0, stream>>>(
      GATED, Wdb, out, H, ROWS, D_MODEL, D_FF);
}
